// CausalSelfAttention_87763361726648
// MI455X (gfx1250) — compile-verified, matrix operand order checked by reading, not run on hardware
//
#include <hip/hip_runtime.h>

#define Bdim 4
#define Lseq 2048
#define Dmod 1024
#define Hn   16
#define DK   64
#define Mrows (Bdim*Lseq)      // 8192
#define N3D  (3*Dmod)          // 3072

typedef __bf16 bf16_t;
typedef __attribute__((ext_vector_type(8)))  __bf16 v8bf;
typedef __attribute__((ext_vector_type(16))) __bf16 v16bf;
typedef __attribute__((ext_vector_type(8)))  float  v8f;
typedef __attribute__((ext_vector_type(4)))  unsigned int v4u;
typedef __attribute__((ext_vector_type(8)))  int v8i;
typedef __attribute__((ext_vector_type(4)))  int v4i;

#if defined(__has_builtin)
#  if __has_builtin(__builtin_amdgcn_tensor_load_to_lds) && __has_builtin(__builtin_amdgcn_s_wait_tensorcnt)
#    define HAVE_TDM 1
#  endif
#endif
#ifndef HAVE_TDM
#  define HAVE_TDM 0
#endif

union FragU { v16bf v; v8bf h[2]; };

// 16x32 A/B fragment for this lane; ISA 16-bit layout:
// elems 0..7 = K(kbase+8h .. +7), elems 8..15 = K(kbase+16+8h .. +7).
__device__ __forceinline__ v16bf load_frag(const bf16_t* rowptr, int kbase, int half) {
  FragU f;
  f.h[0] = *(const v8bf*)(rowptr + kbase + 8 * half);
  f.h[1] = *(const v8bf*)(rowptr + kbase + 16 + 8 * half);
  return f.v;
}

__device__ __forceinline__ v8f wmma_bf16(v16bf a, v16bf b, v8f c) {
  return __builtin_amdgcn_wmma_f32_16x16x32_bf16(false, a, false, b, (short)0, c, false, false);
}

#if HAVE_TDM
// TDM: DMA a contiguous 8KB (4096 x bf16) block from global to LDS.
// D# per CDNA5 ISA 8.3/8.4: 1-row tile, tile_dim0 = 4096, data_size = 2B.
// 6-arg builtin on this toolchain: (g0, g1, g2, g3, g4, cpol).
__device__ __forceinline__ void tdm_load_contig_8k(const bf16_t* gsrc, bf16_t* ldsdst) {
  unsigned long long ga = (unsigned long long)gsrc;
  unsigned int la =
      (unsigned int)(__SIZE_TYPE__)(__attribute__((address_space(3))) bf16_t*)ldsdst;
  v4u g0;
  g0[0] = 1u;                                              // count=1 (valid user D#)
  g0[1] = la;                                              // lds_addr (bytes)
  g0[2] = (unsigned int)(ga & 0xffffffffull);              // global_addr[31:0]
  g0[3] = (unsigned int)((ga >> 32) & 0x1ffffffull)        // global_addr[56:32]
        | (2u << 30);                                      // type=2 ("image")
  v8i g1;
  g1[0] = 0x00010000;                                      // data_size=1 (2 bytes)
  g1[1] = (int)(4096u << 16);                              // tensor_dim0 = 4096 (lo16)
  g1[2] = (int)(1u << 16);                                 // tensor_dim1 = 1 (lo16)
  g1[3] = (int)(4096u << 16);                              // tile_dim0 = 4096
  g1[4] = 0;                                               // tile_dim1 = tile_dim2 = 0
  g1[5] = 4096;                                            // tensor_dim0_stride (lo32)
  g1[6] = 0;
  g1[7] = 0;
  v4i gz4 = {0, 0, 0, 0};
  v8i gz8 = {0, 0, 0, 0, 0, 0, 0, 0};
  __builtin_amdgcn_tensor_load_to_lds(g0, g1, gz4, gz4, gz8, 0);
}
#endif

// ---------------- conversion kernels ----------------
__global__ void cvt_f32_to_bf16(const float* __restrict__ src, bf16_t* __restrict__ dst, int n) {
  int i = blockIdx.x * blockDim.x + threadIdx.x;
  if (i < n) dst[i] = (bf16_t)src[i];
}

// dst[c][r] = src[r][c]
__global__ void transpose_to_bf16(const float* __restrict__ src, bf16_t* __restrict__ dst,
                                  int rows, int cols) {
  int i = blockIdx.x * blockDim.x + threadIdx.x;
  if (i < rows * cols) {
    int c = i / rows;
    int r = i % rows;
    dst[i] = (bf16_t)src[(size_t)r * cols + c];
  }
}

// ---------------- QKV projection GEMM (fully unrolled k) ----------------
__global__ void __launch_bounds__(256)
qkv_gemm(const bf16_t* __restrict__ xb, const bf16_t* __restrict__ WT,
         const float* __restrict__ bias,
         bf16_t* __restrict__ Qb, bf16_t* __restrict__ Kb, bf16_t* __restrict__ Vb) {
  const int m0 = blockIdx.x * 128;
  const int n0 = blockIdx.y * 64;
  const int tid = threadIdx.x;
  const int wave = tid >> 5;
  const int lane = tid & 31;
  const int ln = lane & 15;
  const int half = lane >> 4;

  const int mr = m0 + wave * 16;
  const bf16_t* arow = xb + (size_t)(mr + ln) * Dmod;
  const bf16_t* brow[4];
#pragma unroll
  for (int t = 0; t < 4; ++t) brow[t] = WT + (size_t)(n0 + t * 16 + ln) * Dmod;

  v8f acc[4] = {};
#pragma unroll
  for (int k0 = 0; k0 < Dmod; k0 += 32) {
    v16bf af = load_frag(arow, k0, half);
#pragma unroll
    for (int t = 0; t < 4; ++t) {
      v16bf bf = load_frag(brow[t], k0, half);
      acc[t] = wmma_bf16(af, bf, acc[t]);
    }
  }

#pragma unroll
  for (int t = 0; t < 4; ++t) {
    const int n = n0 + t * 16 + ln;
    const int which = n >> 10;          // 0=Q 1=K 2=V
    const int rem = n & 1023;
    const int head = rem >> 6;
    const int dk = rem & 63;
    const float bv = bias[n];
#pragma unroll
    for (int r = 0; r < 8; ++r) {
      const int row = mr + r + 8 * half;           // global M row = b*L + l
      const int bb = row >> 11;
      const int l = row & 2047;
      const size_t dst = (((size_t)bb * Hn + head) * Lseq + l) * DK + dk;
      float v = acc[t][r] + bv;
      if (which == 0)      Qb[dst] = (bf16_t)(v * 0.125f);   // fold 1/sqrt(64)
      else if (which == 1) Kb[dst] = (bf16_t)v;
      else                 Vb[dst] = (bf16_t)v;
    }
  }
}

// ---------------- flash attention (TDM double-buffered K/V tiles) ----------------
__global__ void __launch_bounds__(256)
flash_attn(const bf16_t* __restrict__ Qb, const bf16_t* __restrict__ Kb,
           const bf16_t* __restrict__ Vb, bf16_t* __restrict__ Ob) {
  const int bh = blockIdx.y;
  const int b = bh / Hn, hh = bh % Hn;
  const int q0 = blockIdx.x * 128;
  const int tid = threadIdx.x;
  const int wave = tid >> 5;
  const int lane = tid & 31;
  const int ln = lane & 15;
  const int half = lane >> 4;

  const bf16_t* Qbase = Qb + ((size_t)bh * Lseq) * DK;
  const bf16_t* Kbase = Kb + ((size_t)bh * Lseq) * DK;
  const bf16_t* Vbase = Vb + ((size_t)bh * Lseq) * DK;

  const int qr = q0 + wave * 16;   // wave's first query row

  const bf16_t* qrow = Qbase + (size_t)(qr + ln) * DK;
  v16bf qf0 = load_frag(qrow, 0, half);
  v16bf qf1 = load_frag(qrow, 32, half);

  v8f o[4] = {};
  float mrow[8], lrow[8];
#pragma unroll
  for (int r = 0; r < 8; ++r) { mrow[r] = -1e30f; lrow[r] = 0.0f; }

  __shared__ __align__(16) bf16_t lds_k[2][64 * 64];    // K tiles, row-major (B-frag ready)
  __shared__ __align__(16) bf16_t lds_vt[2][64 * 64];   // V^T tiles (d-major)
  __shared__ __align__(16) bf16_t lds_p[8 * 16 * 64];   // per-wave P strips
  bf16_t* myp = lds_p + wave * 16 * 64;

  auto stageK = [&](int j0, int buf) {
#if HAVE_TDM
    if (wave == 0) tdm_load_contig_8k(Kbase + (size_t)j0 * DK, &lds_k[buf][0]);
#else
    const bf16_t* Ktile = Kbase + (size_t)j0 * DK;
#pragma unroll
    for (int c = 0; c < 2; ++c) {
      const int off = (tid + c * 256) * 8;
      *(v8bf*)(&lds_k[buf][0] + off) = *(const v8bf*)(Ktile + off);
    }
#endif
  };
  auto stageV = [&](int j0, int buf) {
    const int e0 = tid * 16;
    const int k = e0 >> 6;
    const int d0 = e0 & 63;
    const bf16_t* vrow = Vbase + (size_t)(j0 + k) * DK + d0;
    v8bf va = *(const v8bf*)(vrow);
    v8bf vb = *(const v8bf*)(vrow + 8);
#pragma unroll
    for (int i = 0; i < 8; ++i) {
      lds_vt[buf][(d0 + i) * 64 + k] = va[i];
      lds_vt[buf][(d0 + 8 + i) * 64 + k] = vb[i];
    }
  };

  const int jt_end = (q0 + 127) / 64;
  stageK(0, 0);
  stageV(0, 0);

  for (int jt = 0; jt <= jt_end; ++jt) {
    const int j0 = jt * 64;
    const int cur = jt & 1;

#if HAVE_TDM
    if (wave == 0) __builtin_amdgcn_s_wait_tensorcnt(0);  // K tile jt landed
#endif
    __syncthreads();   // publish lds_k[cur] / lds_vt[cur]

    // kick off staging of tile jt+1 into the other buffer; DMA/loads overlap compute
    if (jt < jt_end) {
      stageK(j0 + 64, cur ^ 1);
      stageV(j0 + 64, cur ^ 1);
      if (j0 + 128 < Lseq) {
        __builtin_prefetch(Kbase + (size_t)(j0 + 128) * DK, 0, 0);
        __builtin_prefetch(Vbase + (size_t)(j0 + 128) * DK, 0, 0);
      }
    }

    if (j0 <= qr + 15) {   // wave-uniform: skip fully-masked key tiles
      // S = Q @ K^T  (4 column subtiles of 16 keys); K rows straight from LDS
      v8f s[4];
#pragma unroll
      for (int t = 0; t < 4; ++t) {
        const bf16_t* krow = &lds_k[cur][0] + (t * 16 + ln) * DK;
        v16bf kf0 = load_frag(krow, 0, half);
        v16bf kf1 = load_frag(krow, 32, half);
        v8f a = {};
        a = wmma_bf16(qf0, kf0, a);
        a = wmma_bf16(qf1, kf1, a);
        s[t] = a;
      }

      // online softmax per row (rows live in VGPR r, lanes 0-15 / 16-31)
      float alpha[8];
#pragma unroll
      for (int r = 0; r < 8; ++r) {
        const int row = qr + r + 8 * half;
        float tm = -1e30f;
#pragma unroll
        for (int t = 0; t < 4; ++t) {
          const int col = j0 + t * 16 + ln;
          float v = s[t][r];
          if (col > row) v = -1e30f;
          s[t][r] = v;
          tm = fmaxf(tm, v);
        }
#pragma unroll
        for (int m = 1; m < 16; m <<= 1) tm = fmaxf(tm, __shfl_xor(tm, m, 16));
        const float mnew = fmaxf(mrow[r], tm);
        alpha[r] = __expf(mrow[r] - mnew);
        float rs = 0.0f;
#pragma unroll
        for (int t = 0; t < 4; ++t) {
          float p = __expf(s[t][r] - mnew);
          rs += p;
          myp[(r + 8 * half) * 64 + t * 16 + ln] = (bf16_t)p;
        }
#pragma unroll
        for (int m = 1; m < 16; m <<= 1) rs += __shfl_xor(rs, m, 16);
        lrow[r] = lrow[r] * alpha[r] + rs;
        mrow[r] = mnew;
      }

      // P back in A-fragment layout via wave-private LDS strip
      const bf16_t* prow = myp + ln * 64;
      v16bf pf0 = load_frag(prow, 0, half);
      v16bf pf1 = load_frag(prow, 32, half);

      // O = diag(alpha)*O + P @ V
#pragma unroll
      for (int t = 0; t < 4; ++t) {
        v8f a = o[t];
#pragma unroll
        for (int r = 0; r < 8; ++r) a[r] *= alpha[r];
        const bf16_t* vtrow = &lds_vt[cur][0] + (t * 16 + ln) * 64;
        v16bf vf0 = load_frag(vtrow, 0, half);
        v16bf vf1 = load_frag(vtrow, 32, half);
        a = wmma_bf16(pf0, vf0, a);
        a = wmma_bf16(pf1, vf1, a);
        o[t] = a;
      }
    }
    __syncthreads();   // all reads of buffers done before next overwrite
  }

  // normalize and scatter to [B, L, D] bf16 for the output GEMM
  float inv[8];
#pragma unroll
  for (int r = 0; r < 8; ++r) inv[r] = 1.0f / lrow[r];
#pragma unroll
  for (int t = 0; t < 4; ++t) {
    const int d = t * 16 + ln;
#pragma unroll
    for (int r = 0; r < 8; ++r) {
      const int l = qr + r + 8 * half;
      Ob[((size_t)b * Lseq + l) * Dmod + hh * DK + d] = (bf16_t)(o[t][r] * inv[r]);
    }
  }
}

// ---------------- output projection GEMM (fully unrolled k) ----------------
__global__ void __launch_bounds__(256)
out_gemm(const bf16_t* __restrict__ ob, const bf16_t* __restrict__ WT,
         const float* __restrict__ bias, float* __restrict__ out) {
  const int m0 = blockIdx.x * 128;
  const int n0 = blockIdx.y * 64;
  const int tid = threadIdx.x;
  const int wave = tid >> 5;
  const int lane = tid & 31;
  const int ln = lane & 15;
  const int half = lane >> 4;

  const int mr = m0 + wave * 16;
  const bf16_t* arow = ob + (size_t)(mr + ln) * Dmod;
  const bf16_t* brow[4];
#pragma unroll
  for (int t = 0; t < 4; ++t) brow[t] = WT + (size_t)(n0 + t * 16 + ln) * Dmod;

  v8f acc[4] = {};
#pragma unroll
  for (int k0 = 0; k0 < Dmod; k0 += 32) {
    v16bf af = load_frag(arow, k0, half);
#pragma unroll
    for (int t = 0; t < 4; ++t) {
      v16bf bf = load_frag(brow[t], k0, half);
      acc[t] = wmma_bf16(af, bf, acc[t]);
    }
  }

#pragma unroll
  for (int t = 0; t < 4; ++t) {
    const int n = n0 + t * 16 + ln;
    const float bv = bias[n];
#pragma unroll
    for (int r = 0; r < 8; ++r) {
      const int row = mr + r + 8 * half;
      out[(size_t)row * Dmod + n] = acc[t][r] + bv;
    }
  }
}

// ---------------- launch ----------------
extern "C" void kernel_launch(void* const* d_in, const int* in_sizes, int n_in,
                              void* d_out, int out_size, void* d_ws, size_t ws_size,
                              hipStream_t stream) {
  const float* x     = (const float*)d_in[0];
  const float* W_qkv = (const float*)d_in[1];
  const float* b_qkv = (const float*)d_in[2];
  const float* W_out = (const float*)d_in[3];
  const float* b_out = (const float*)d_in[4];
  float* out = (float*)d_out;

  char* ws = (char*)d_ws;
  size_t off = 0;
  bf16_t* xb    = (bf16_t*)(ws + off); off += (size_t)Mrows * Dmod * 2;
  bf16_t* WqkvT = (bf16_t*)(ws + off); off += (size_t)N3D * Dmod * 2;
  bf16_t* WoutT = (bf16_t*)(ws + off); off += (size_t)Dmod * Dmod * 2;
  bf16_t* Qb    = (bf16_t*)(ws + off); off += (size_t)Bdim * Hn * Lseq * DK * 2;
  bf16_t* Kb    = (bf16_t*)(ws + off); off += (size_t)Bdim * Hn * Lseq * DK * 2;
  bf16_t* Vb    = (bf16_t*)(ws + off); off += (size_t)Bdim * Hn * Lseq * DK * 2;
  bf16_t* Ob    = (bf16_t*)(ws + off); off += (size_t)Mrows * Dmod * 2;
  (void)ws_size; (void)in_sizes; (void)n_in; (void)out_size;

  const int nx = Mrows * Dmod;
  cvt_f32_to_bf16<<<(nx + 255) / 256, 256, 0, stream>>>(x, xb, nx);
  const int nwq = Dmod * N3D;
  transpose_to_bf16<<<(nwq + 255) / 256, 256, 0, stream>>>(W_qkv, WqkvT, Dmod, N3D);
  const int nwo = Dmod * Dmod;
  transpose_to_bf16<<<(nwo + 255) / 256, 256, 0, stream>>>(W_out, WoutT, Dmod, Dmod);

  qkv_gemm<<<dim3(Mrows / 128, N3D / 64), 256, 0, stream>>>(xb, WqkvT, b_qkv, Qb, Kb, Vb);
  flash_attn<<<dim3(Lseq / 128, Bdim * Hn), 256, 0, stream>>>(Qb, Kb, Vb, Ob);
  out_gemm<<<dim3(Mrows / 128, Dmod / 64), 256, 0, stream>>>(Ob, WoutT, b_out, out);
}
